// STKernelAttentionBlock_55997783605735
// MI455X (gfx1250) — matrix operand order repaired and checked by reading, not compiled, not run on hardware
//
#include <hip/hip_runtime.h>
#include <cstdint>
#include <cstddef>

#define Nn 64
#define Cc 128
#define Tt 128
#define Vv 25
#define Ss 3
#define Kk 7

typedef float  v2f   __attribute__((ext_vector_type(2)));
typedef float  v8f   __attribute__((ext_vector_type(8)));
typedef __bf16 v16bf __attribute__((ext_vector_type(16)));
typedef unsigned int v4u __attribute__((ext_vector_type(4)));
typedef int    v8i   __attribute__((ext_vector_type(8)));
typedef int    v4i   __attribute__((ext_vector_type(4)));
typedef unsigned int  u32;
typedef unsigned short u16;

// ---------------- workspace layout (bytes) ----------------
// G     : 64*625*4        = 160000
// maxd2 : 4
// att   : 64*3*625*4      = 480000
// Wpk   : 7*3*64*128*4    = 688128   (bf16 pairs packed as u32, [k][s][c/2][o])
// vx    : 64*128*128*25*4 = 104857600
#define G_OFF    0u
#define MAX_OFF  160000u
#define ATT_OFF  160256u
#define WPK_OFF  640256u
#define VX_OFF   2016768u
// total ~ 106.9 MB

__device__ __forceinline__ u16 f2bf(float f) {
  u32 u = __float_as_uint(f);
  u += ((u >> 16) & 1u) + 0x7FFFu;   // round-to-nearest-even
  return (u16)(u >> 16);
}

union BFV { u32 u[8]; v16bf v; };

// ---------------------------------------------------------------------------
// TDM: 2-D tile load global -> LDS (Tensor Data Mover, ISA ch.8).
// D# group0: count=1, lds_addr, 57-bit global addr, type=2 ("image").
// D# group1: data_size=4B, huge tensor dims (no OOB clip), tile_d0 x tile_d1,
//            line stride in elements. Groups 2/3 zero (2-D tensor).
// ---------------------------------------------------------------------------
__device__ __forceinline__ void tdm_load_2d(void* lds_ptr, const void* gptr,
                                            unsigned tile_d0, unsigned tile_d1,
                                            unsigned long long stride0_elems) {
  unsigned long long ga = (unsigned long long)(size_t)gptr;
  unsigned lds_off = (unsigned)(size_t)lds_ptr;      // low 32 bits = LDS offset
  const unsigned td0 = 0x40000000u, td1 = 0x40000000u;
  v4u g0 = {0u, 0u, 0u, 0u};
  g0[0] = 1u;                                        // count=1, user mode
  g0[1] = lds_off;
  g0[2] = (unsigned)ga;
  g0[3] = (unsigned)((ga >> 32) & 0x01FFFFFFull) | (2u << 30);
  v8i g1 = {0, 0, 0, 0, 0, 0, 0, 0};
  g1[0] = (int)(2u << 16);                           // data_size = 4 bytes
  g1[1] = (int)((td0 & 0xFFFFu) << 16);              // tensor_dim0 lo
  g1[2] = (int)((td0 >> 16) | ((td1 & 0xFFFFu) << 16));
  g1[3] = (int)((td1 >> 16) | (tile_d0 << 16));      // tile_dim0
  g1[4] = (int)(tile_d1 & 0xFFFFu);                  // tile_dim1 (tile_dim2=0)
  g1[5] = (int)(unsigned)(stride0_elems & 0xFFFFFFFFull);
  g1[6] = (int)(unsigned)((stride0_elems >> 32) & 0xFFFFull);
  v4i gz = {0, 0, 0, 0};
#if __has_include(<hip/amd_detail/amd_gfx1250_TDM.h>)
  v8i z8 = {0, 0, 0, 0, 0, 0, 0, 0};
  __builtin_amdgcn_tensor_load_to_lds(g0, g1, gz, gz, z8, 0);
#else
  __builtin_amdgcn_tensor_load_to_lds(g0, g1, gz, gz, 0);
#endif
}

// ---------------------------------------------------------------------------
// K0: pack out_w (f32 [o][sc][k]) -> bf16 pairs u32 [k][s][cp][o]
// ---------------------------------------------------------------------------
__global__ void k_pack_w(const float* __restrict__ w, u32* __restrict__ wp) {
  int idx = blockIdx.x * 256 + threadIdx.x;          // 7*3*64*128 = 172032
  if (idx >= Kk * Ss * 64 * Cc) return;
  int o    = idx & 127;
  int cp   = (idx >> 7) & 63;
  int rest = idx >> 13;                              // k*3+s  (64*128 = 2^13)
  int s = rest % 3, k = rest / 3;
  int sc0 = s * Cc + 2 * cp;
  float lo = w[(o * (Ss * Cc) + sc0) * Kk + k];
  float hi = w[(o * (Ss * Cc) + sc0 + 1) * Kk + k];
  wp[idx] = (u32)f2bf(lo) | ((u32)f2bf(hi) << 16);
}

// ---------------------------------------------------------------------------
// K1: per-n Gram of yf rows (y = x + posenc).  G[n][u][v] = <yf_u, yf_v>
// ---------------------------------------------------------------------------
__global__ void k_gram(const float* __restrict__ x, float* __restrict__ G) {
  __shared__ float pe[Cc][Vv];
  __shared__ float tile[128][Vv];
  int n = blockIdx.x, tid = threadIdx.x;

  for (int idx = tid; idx < Cc * Vv; idx += 256) {
    int c = idx / Vv, v = idx % Vv;
    float div = expf(-(float)(c & ~1) * (logf(10000.0f) / (float)Cc));
    float ang = (float)v * div;
    pe[c][v] = (c & 1) ? cosf(ang) : sinf(ang);
  }

  int u0 = 0, v0 = 0, u1 = -1, v1 = -1;              // 325 (u<=v) pairs
  {
    int p = tid, u = 0, base = 0;
    while (p >= base + (Vv - u)) { base += Vv - u; ++u; }
    u0 = u; v0 = u + (p - base);
  }
  if (tid + 256 < 325) {
    int p = tid + 256, u = 0, base = 0;
    while (p >= base + (Vv - u)) { base += Vv - u; ++u; }
    u1 = u; v1 = u + (p - base);
  }
  float a0 = 0.f, a1 = 0.f;
  __syncthreads();

  const float* xn = x + (size_t)n * Cc * Tt * Vv;
  for (int ct0 = 0; ct0 < Cc * Tt; ct0 += 128) {
    for (int idx = tid; idx < 128 * Vv; idx += 256) {
      int r = idx / Vv, v = idx % Vv;
      int ct = ct0 + r;
      int c = ct >> 7;                               // T = 128
      tile[r][v] = xn[(size_t)ct * Vv + v] + pe[c][v];
    }
    __syncthreads();
    for (int r = 0; r < 128; ++r) {
      a0 = fmaf(tile[r][u0], tile[r][v0], a0);
      if (u1 >= 0) a1 = fmaf(tile[r][u1], tile[r][v1], a1);
    }
    __syncthreads();
  }
  float* Gn = G + n * Vv * Vv;
  Gn[u0 * Vv + v0] = a0; Gn[v0 * Vv + u0] = a0;
  if (u1 >= 0) { Gn[u1 * Vv + v1] = a1; Gn[v1 * Vv + u1] = a1; }
}

// ---------------------------------------------------------------------------
// K2: max over d2 (dism normalization: (C*T)^2 cancels -> dism = d2/max(d2))
// ---------------------------------------------------------------------------
__global__ void k_maxd2(const float* __restrict__ G, float* __restrict__ mout) {
  __shared__ float red[256];
  int tid = threadIdx.x;
  float m = 0.f;
  for (int idx = tid; idx < Nn * Vv * Vv; idx += 256) {
    int n = idx / (Vv * Vv), r = idx % (Vv * Vv);
    int u = r / Vv, v = r % Vv;
    const float* Gn = G + n * Vv * Vv;
    float d2 = Gn[u * Vv + u] + Gn[v * Vv + v] - 2.f * Gn[u * Vv + v];
    m = fmaxf(m, fmaxf(d2, 0.f));
  }
  red[tid] = m; __syncthreads();
  for (int s = 128; s > 0; s >>= 1) {
    if (tid < s) red[tid] = fmaxf(red[tid], red[tid + s]);
    __syncthreads();
  }
  if (tid == 0) mout[0] = red[0];
}

// ---------------------------------------------------------------------------
// K3: kernel attention + sparsemax (axis=u) + atts/alphas combine
// ---------------------------------------------------------------------------
__global__ void k_att(const float* __restrict__ G, const float* __restrict__ maxp,
                      const float* __restrict__ theta, const float* __restrict__ atts,
                      const float* __restrict__ alphas, float* __restrict__ att) {
  int n = blockIdx.x, tid = threadIdx.x;
  if (tid >= Ss * Vv) return;
  int s = tid / Vv, vc = tid % Vv;
  const float* Gn = G + n * Vv * Vv;
  float inv_max = 1.f / maxp[0];
  float th = expf(theta[s]);
  float z[Vv], zs[Vv];
  float gvv = Gn[vc * Vv + vc];
  for (int u = 0; u < Vv; ++u) {
    float d2 = Gn[u * Vv + u] + gvv - 2.f * Gn[u * Vv + vc];
    d2 = fmaxf(d2, 0.f);
    z[u] = expf(-th * d2 * inv_max);
    zs[u] = z[u];
  }
  for (int i = 1; i < Vv; ++i) {                     // sort descending
    float key = zs[i]; int j = i - 1;
    while (j >= 0 && zs[j] < key) { zs[j + 1] = zs[j]; --j; }
    zs[j + 1] = key;
  }
  float cs = 0.f, csk = 1.f; int ks = 1;
  for (int k = 1; k <= Vv; ++k) {
    cs += zs[k - 1];
    if (1.f + (float)k * zs[k - 1] > cs) { ks = k; csk = cs; }
  }
  float tau = (csk - 1.f) / (float)ks;
  float al = alphas[s];
  float* an = att + (size_t)((n * Ss + s) * Vv) * Vv;
  for (int u = 0; u < Vv; ++u) {
    float sm = fmaxf(z[u] - tau, 0.f);
    an[u * Vv + vc] = atts[(s * Vv + u) * Vv + vc] + sm * al;
  }
}

// ---------------------------------------------------------------------------
// K4/K6: 1x1 conv GEMM via V_WMMA_F32_16X16X4_F32.
// mode 0: dst = W@src + bb                     (value map -> vx)
// mode 1: dst = leaky(xres + BN(W@src + bb))   (ff net, in-place on d_out)
// block = (t-tile of 16, n); 256 threads = 8 waves; dyn LDS tile[128][400]
// Tile staged by TDM (tensor_load_to_lds); A fragments register-resident,
// B fragments double-buffered so WMMA overlaps LDS latency.
// ---------------------------------------------------------------------------
__global__ void k_gemm1x1(const float* __restrict__ src, const float* __restrict__ W,
                          const float* __restrict__ bb, const float* __restrict__ gg,
                          const float* __restrict__ mm, const float* __restrict__ vvv,
                          const float* __restrict__ bbe, const float* __restrict__ xres,
                          float* __restrict__ dst, int mode) {
  extern __shared__ float tile[];                    // [128][400]
  int n = blockIdx.y, t0 = blockIdx.x * 16, tid = threadIdx.x;
  const float* sn = src + (size_t)n * Cc * Tt * Vv;

  if ((tid >> 5) == 0) {                             // wave 0 drives the TDM
    tdm_load_2d(tile, sn + (size_t)t0 * Vv, 400u, 128u,
                (unsigned long long)(Tt * Vv));
    __builtin_amdgcn_s_wait_tensorcnt(0);
  }
  __syncthreads();

  int wave = tid >> 5, lane = tid & 31;
  int m = lane & 15;
  int kb = (lane < 16) ? 0 : 2;                      // f32 A/B lane K-offset
  int o0 = wave * 16;

  v2f Areg[32];                                      // whole A column per wave
#pragma unroll
  for (int i = 0; i < 32; ++i)
    Areg[i] = *(const v2f*)(W + (o0 + m) * Cc + 4 * i + kb);

  for (int cb = 0; cb < 25; ++cb) {
    int col0 = cb * 16;
    v8f acc = {0.f, 0.f, 0.f, 0.f, 0.f, 0.f, 0.f, 0.f};
    v2f B0;
    B0.x = tile[(kb + 0) * 400 + col0 + m];
    B0.y = tile[(kb + 1) * 400 + col0 + m];
#pragma unroll
    for (int i = 0; i < 32; ++i) {
      v2f B1;
      if (i < 31) {                                  // prefetch next B frag
        B1.x = tile[((i + 1) * 4 + kb + 0) * 400 + col0 + m];
        B1.y = tile[((i + 1) * 4 + kb + 1) * 400 + col0 + m];
      }
      acc = __builtin_amdgcn_wmma_f32_16x16x4_f32(false, Areg[i], false, B0,
                                                  (short)0, acc, false, false);
      B0 = B1;
    }
    for (int r = 0; r < 8; ++r) {
      int o = o0 + r + ((lane < 16) ? 0 : 8);
      int col = col0 + m;
      size_t off = ((size_t)(n * Cc + o) * Tt + t0) * Vv + col;
      float val;
      if (mode == 0) {
        val = acc[r] + bb[o];
      } else {
        float inv = gg[o] * rsqrtf(vvv[o] + 1e-5f);
        float y = xres[off] + acc[r] * inv + (bb[o] - mm[o]) * inv + bbe[o];
        val = (y > 0.f) ? y : 0.1f * y;
      }
      dst[off] = val;
    }
  }
}

// ---------------------------------------------------------------------------
// K5: fused attention-apply + temporal conv(K=7) + BN + residual + leaky.
// block = (t-tile16, v, n); 256 threads = 8 waves (one 16-row o-chunk each).
// z[s][t'][c] built in LDS (bf16); GEMM via V_WMMA_F32_16X16X32_BF16,
// 84 K-steps software-pipelined (A global / B LDS double-buffered).
// ---------------------------------------------------------------------------
__global__ void k_conv(const float* __restrict__ vx, const float* __restrict__ att,
                       const u32* __restrict__ Wpk,
                       const float* __restrict__ ob, const float* __restrict__ og,
                       const float* __restrict__ om, const float* __restrict__ ov,
                       const float* __restrict__ obe,
                       const float* __restrict__ x, float* __restrict__ hout) {
  __shared__ u16 zl[Ss][22][Cc];                     // bf16 z tile, t' = t0-3..t0+18
  __shared__ float attL[Ss][Vv];
  int t0 = blockIdx.x * 16, v = blockIdx.y, n = blockIdx.z;
  int tid = threadIdx.x;

  if (tid < Ss * Vv) {
    int s = tid / Vv, u = tid % Vv;
    attL[s][u] = att[(size_t)((n * Ss + s) * Vv + u) * Vv + v];
  }
  __syncthreads();

  for (int idx = tid; idx < 22 * Cc; idx += 256) {   // phase 1: z = vx @ att
    int tp = idx / Cc, c = idx % Cc;
    int t = t0 - 3 + tp;
    float a0 = 0.f, a1 = 0.f, a2 = 0.f;
    if (t >= 0 && t < Tt) {
      const float* vp = vx + ((size_t)(n * Cc + c) * Tt + t) * Vv;
#pragma unroll
      for (int u = 0; u < Vv; ++u) {
        float w = vp[u];
        a0 = fmaf(w, attL[0][u], a0);
        a1 = fmaf(w, attL[1][u], a1);
        a2 = fmaf(w, attL[2][u], a2);
      }
    }
    zl[0][tp][c] = f2bf(a0);
    zl[1][tp][c] = f2bf(a1);
    zl[2][tp][c] = f2bf(a2);
  }
  __syncthreads();

  int wave = tid >> 5, lane = tid & 31;
  int o0 = wave * 16;
  int m = lane & 15;
  int hi = (lane < 16) ? 0 : 1;

  // fragment loader for linear step st = (k*Ss + s)*4 + cb
  auto load_frag = [&](int st, u32* a, u32* b) {
    int k  = st / 12;
    int r2 = st - k * 12;
    int s  = r2 >> 2;
    int cb = r2 & 3;
    int c0 = cb * 32;
    const u32* wbase = Wpk + (size_t)((k * Ss + s) * 64) * Cc;  // [cp][o]
    const u16* zrow = &zl[s][m + k][0];
#pragma unroll
    for (int r = 0; r < 4; ++r) {                    // A: 16x32 bf16 layout
      int kd = hi * 8 + 2 * r;
      a[r]     = wbase[(size_t)((c0 + kd) >> 1) * Cc + o0 + m];
      a[r + 4] = wbase[(size_t)((c0 + kd + 16) >> 1) * Cc + o0 + m];
    }
#pragma unroll
    for (int r = 0; r < 8; ++r)                      // B: 32x16 bf16 layout
      b[r] = *(const u32*)&zrow[c0 + hi * 8 + 2 * r + (r >> 2) * 8];
  };

  v8f acc = {0.f, 0.f, 0.f, 0.f, 0.f, 0.f, 0.f, 0.f};
  u32 a_cur[8], b_cur[8];
  load_frag(0, a_cur, b_cur);
  for (int st = 0; st < Kk * Ss * 4; ++st) {
    u32 a_nxt[8], b_nxt[8];
    if (st + 1 < Kk * Ss * 4) load_frag(st + 1, a_nxt, b_nxt);
    BFV ua, ub;
#pragma unroll
    for (int r = 0; r < 8; ++r) { ua.u[r] = a_cur[r]; ub.u[r] = b_cur[r]; }
    acc = __builtin_amdgcn_wmma_f32_16x16x32_bf16(false, ua.v, false, ub.v,
                                                  (short)0, acc, false, false);
#pragma unroll
    for (int r = 0; r < 8; ++r) { a_cur[r] = a_nxt[r]; b_cur[r] = b_nxt[r]; }
  }

  for (int r = 0; r < 8; ++r) {                      // BN + residual + leaky
    int o = o0 + r + hi * 8;
    int t = t0 + m;
    float inv = og[o] * rsqrtf(ov[o] + 1e-5f);
    float val = acc[r] * inv + (ob[o] - om[o]) * inv + obe[o];
    size_t off = ((size_t)(n * Cc + o) * Tt + t) * Vv + v;
    float y = x[off] + val;
    hout[off] = (y > 0.f) ? y : 0.1f * y;
  }
}

// ---------------------------------------------------------------------------
extern "C" void kernel_launch(void* const* d_in, const int* in_sizes, int n_in,
                              void* d_out, int out_size, void* d_ws, size_t ws_size,
                              hipStream_t stream) {
  (void)in_sizes; (void)n_in; (void)out_size; (void)ws_size;
  const float* x      = (const float*)d_in[0];
  const float* theta  = (const float*)d_in[1];
  const float* atts   = (const float*)d_in[2];
  const float* alphas = (const float*)d_in[3];
  const float* v_w    = (const float*)d_in[4];
  const float* v_b    = (const float*)d_in[5];
  const float* out_w  = (const float*)d_in[6];
  const float* out_b  = (const float*)d_in[7];
  const float* out_g  = (const float*)d_in[8];
  const float* out_be = (const float*)d_in[9];
  const float* out_m  = (const float*)d_in[10];
  const float* out_v  = (const float*)d_in[11];
  const float* ff_w   = (const float*)d_in[12];
  const float* ff_b   = (const float*)d_in[13];
  const float* ff_g   = (const float*)d_in[14];
  const float* ff_be  = (const float*)d_in[15];
  const float* ff_m   = (const float*)d_in[16];
  const float* ff_v   = (const float*)d_in[17];
  float* out = (float*)d_out;

  char* ws = (char*)d_ws;
  float* G    = (float*)(ws + G_OFF);
  float* maxp = (float*)(ws + MAX_OFF);
  float* att  = (float*)(ws + ATT_OFF);
  u32*   wpk  = (u32*)(ws + WPK_OFF);
  float* vx   = (float*)(ws + VX_OFF);

  k_pack_w<<<(Kk * Ss * 64 * Cc) / 256, 256, 0, stream>>>(out_w, wpk);
  k_gram  <<<Nn, 256, 0, stream>>>(x, G);
  k_maxd2 <<<1, 256, 0, stream>>>(G, maxp);
  k_att   <<<Nn, 128, 0, stream>>>(G, maxp, theta, atts, alphas, att);

  size_t lds = (size_t)Cc * 400 * sizeof(float);     // 204800 B (<= 320 KB/WGP)
  k_gemm1x1<<<dim3(Tt / 16, Nn), 256, lds, stream>>>(
      x, v_w, v_b, nullptr, nullptr, nullptr, nullptr, nullptr, vx, 0);

  k_conv<<<dim3(Tt / 16, Vv, Nn), 256, 0, stream>>>(
      vx, att, wpk, out_b, out_g, out_m, out_v, out_be, x, out);

  k_gemm1x1<<<dim3(Tt / 16, Nn), 256, lds, stream>>>(
      out, ff_w, ff_b, ff_g, ff_m, ff_v, ff_be, x, out, 1);
}
